// GAT_22514218565930
// MI455X (gfx1250) — matrix operand order repaired and checked by reading, not compiled, not run on hardware
//
#include <hip/hip_runtime.h>
#include <cmath>

#define N_NODES 20000
#define N_EDGES 320000
#define HEADS   4
#define DHEAD   128
#define CH      512   // HEADS*DHEAD
#define F_IN    128

typedef __attribute__((ext_vector_type(16))) _Float16 v16h;
typedef __attribute__((ext_vector_type(8)))  float    v8f;

__device__ inline void atomicMaxF(float* addr, float v) {
  // sign-aware float max via integer atomics (valid with -inf init)
  if (v >= 0.f) atomicMax((int*)addr, __float_as_int(v));
  else          atomicMin((unsigned int*)addr, __float_as_uint(v));
}

// ---- W[K,N] f32 -> Wt[N,K] f16 (transpose + convert) ----
__global__ void wt_conv_kernel(const float* __restrict__ W, _Float16* __restrict__ Wt,
                               int K, int N) {
  int idx = blockIdx.x * blockDim.x + threadIdx.x;
  if (idx >= K * N) return;
  int k = idx / N, n = idx - k * N;
  Wt[(size_t)n * K + k] = (_Float16)W[idx];
}

// ---- WMMA GEMM: C[M,N] f32 = A[M,K] f32 x B (given as Bt[N,K] f16) ----
// one wave computes a 16x64 tile: 1 A-fragment reused across 4 WMMAs per K-step
__global__ void gemm_wmma_kernel(const float* __restrict__ A, const _Float16* __restrict__ Bt,
                                 float* __restrict__ C, int M, int N, int K) {
  int lane  = threadIdx.x & 31;
  int gwave = (blockIdx.x * blockDim.x + threadIdx.x) >> 5;
  int ntiles = N >> 6;
  int mtile = gwave / ntiles;
  int nt    = gwave - mtile * ntiles;
  if (mtile * 16 >= M) return;           // whole-wave uniform exit
  int r16 = lane & 15;
  int hi  = lane >> 4;                   // lane group 0/1
  int kgA = hi * 16;                     // A: K 0-15 vs 16-31
  int kgB = hi * 16;                     // B: K 0-15 vs 16-31
  const float* ap0 = A + (size_t)(mtile * 16 + r16) * K;
  v8f c0 = {}, c1 = {}, c2 = {}, c3 = {};
  for (int k0 = 0; k0 < K; k0 += 32) {
    v16h a;
    const float* ap = ap0 + k0 + kgA;
#pragma unroll
    for (int j = 0; j < 8; ++j) {
      a[j]     = (_Float16)ap[j];        // K = k0+kgA+j
      a[j + 8] = (_Float16)ap[8 + j];    // K = k0+kgA+8+j
    }
    v16h b0, b1, b2, b3;
    const _Float16* bp = Bt + (size_t)(nt * 64 + r16) * K + k0 + kgB;
#pragma unroll
    for (int j = 0; j < 16; ++j) b0[j] = bp[j];
    bp += (size_t)16 * K;
#pragma unroll
    for (int j = 0; j < 16; ++j) b1[j] = bp[j];
    bp += (size_t)16 * K;
#pragma unroll
    for (int j = 0; j < 16; ++j) b2[j] = bp[j];
    bp += (size_t)16 * K;
#pragma unroll
    for (int j = 0; j < 16; ++j) b3[j] = bp[j];
    c0 = __builtin_amdgcn_wmma_f32_16x16x32_f16(false, a, false, b0, (short)0, c0, false, false);
    c1 = __builtin_amdgcn_wmma_f32_16x16x32_f16(false, a, false, b1, (short)0, c1, false, false);
    c2 = __builtin_amdgcn_wmma_f32_16x16x32_f16(false, a, false, b2, (short)0, c2, false, false);
    c3 = __builtin_amdgcn_wmma_f32_16x16x32_f16(false, a, false, b3, (short)0, c3, false, false);
  }
  // D layout: element r -> row M = hi*8 + r, col N = lane&15 (per 16x16 tile)
  int row  = mtile * 16 + hi * 8;
  int colb = nt * 64 + r16;
#pragma unroll
  for (int r = 0; r < 8; ++r) {
    float* cp = C + (size_t)(row + r) * N + colb;
    cp[0]  = c0[r];
    cp[16] = c1[r];
    cp[32] = c2[r];
    cp[48] = c3[r];
  }
}

// ---- one wave per (node, head): el = z.al, er = z.ar (coalesced + shfl reduce) ----
__global__ void attn_score_kernel(const float* __restrict__ Z, const float* __restrict__ al,
                                  const float* __restrict__ ar,
                                  float* __restrict__ EL, float* __restrict__ ER) {
  int gwave = (blockIdx.x * blockDim.x + threadIdx.x) >> 5;
  int lane  = threadIdx.x & 31;
  if (gwave >= N_NODES * HEADS) return;
  int n = gwave >> 2, h = gwave & 3;
  const float* z   = Z + (size_t)n * CH + h * DHEAD;
  const float* alp = al + h * DHEAD;
  const float* arp = ar + h * DHEAD;
  float sl = 0.f, sr = 0.f;
#pragma unroll
  for (int j = 0; j < DHEAD / 32; ++j) {
    int d = lane + j * 32;
    float zv = z[d];
    sl += zv * alp[d];
    sr += zv * arp[d];
  }
#pragma unroll
  for (int off = 16; off > 0; off >>= 1) {
    sl += __shfl_down(sl, off);
    sr += __shfl_down(sr, off);
  }
  if (lane == 0) { EL[gwave] = sl; ER[gwave] = sr; }
}

__global__ void fill_kernel(float* p, float v, int n) {
  int i = blockIdx.x * blockDim.x + threadIdx.x;
  if (i < n) p[i] = v;
}

// ---- pass 1: e = leaky_relu(el[src]+er[dst]); segment max over dst ----
__global__ void edge_max_kernel(const int* __restrict__ src, const int* __restrict__ dst,
                                const float* __restrict__ EL, const float* __restrict__ ER,
                                float* __restrict__ EA, float* __restrict__ EM) {
  int idx = blockIdx.x * blockDim.x + threadIdx.x;
  if (idx >= N_EDGES * HEADS) return;
  int e = idx >> 2, h = idx & 3;
  int s = src[e], d = dst[e];
  float v = EL[s * HEADS + h] + ER[d * HEADS + h];
  v = (v > 0.f) ? v : 0.2f * v;
  EA[idx] = v;
  atomicMaxF(&EM[d * HEADS + h], v);
}

// ---- pass 2: a = exp(e - emax[dst]); segment sum denom ----
__global__ void edge_exp_kernel(const int* __restrict__ dst, float* __restrict__ EA,
                                const float* __restrict__ EM, float* __restrict__ DEN) {
  int idx = blockIdx.x * blockDim.x + threadIdx.x;
  if (idx >= N_EDGES * HEADS) return;
  int e = idx >> 2, h = idx & 3;
  int d = dst[e];
  float a = __expf(EA[idx] - EM[d * HEADS + h]);
  EA[idx] = a;
  atomicAdd(&DEN[d * HEADS + h], a);
}

// ---- pass 3: out[dst] += (a/denom) * z[src] ----
// one wave per EDGE, covering all 4 heads (512 contiguous floats = full
// cacheline-coalesced gather + atomics); head index dd>>7 is constant per
// unrolled j, so alpha selection is register-only.
__global__ void edge_scatter_kernel(const int* __restrict__ src, const int* __restrict__ dst,
                                    const float* __restrict__ EA, const float* __restrict__ DEN,
                                    const float* __restrict__ Z, float* __restrict__ ACC) {
  int gwave = (blockIdx.x * blockDim.x + threadIdx.x) >> 5;
  int lane  = threadIdx.x & 31;
  if (gwave >= N_EDGES) return;
  int e = gwave;
  int s = src[e], d = dst[e];
  float alpha[HEADS];
#pragma unroll
  for (int h = 0; h < HEADS; ++h)
    alpha[h] = EA[e * HEADS + h] / fmaxf(DEN[d * HEADS + h], 1e-9f);
  const float* zs = Z + (size_t)s * CH;
  float*       op = ACC + (size_t)d * CH;
#pragma unroll
  for (int j = 0; j < CH / 32; ++j) {
    int dd = lane + j * 32;
    atomicAdd(&op[dd], alpha[j >> 2] * zs[dd]);   // head = (j*32)>>7 = j>>2
  }
}

// ---- h = elu(acc + b), in place ----
__global__ void bias_elu_kernel(float* __restrict__ ACC, const float* __restrict__ b) {
  int idx = blockIdx.x * blockDim.x + threadIdx.x;
  if (idx >= N_NODES * CH) return;
  int c = idx & (CH - 1);
  float x = ACC[idx] + b[c];
  ACC[idx] = (x > 0.f) ? x : (__expf(x) - 1.f);
}

// ---- readout: one wave per node: sum_n ( mean_h(acc + b2) . Wr + br ) ----
__global__ void readout_kernel(const float* __restrict__ ACC, const float* __restrict__ b2,
                               const float* __restrict__ Wr, const float* __restrict__ br,
                               float* __restrict__ out) {
  int gwave = (blockIdx.x * blockDim.x + threadIdx.x) >> 5;
  int lane  = threadIdx.x & 31;
  if (gwave >= N_NODES) return;
  const float* a = ACC + (size_t)gwave * CH;
  float p = 0.f;
#pragma unroll
  for (int j = 0; j < DHEAD / 32; ++j) {
    int d = lane + j * 32;
    float m = (a[d] + b2[d]) + (a[d + 128] + b2[d + 128]) +
              (a[d + 256] + b2[d + 256]) + (a[d + 384] + b2[d + 384]);
    p += 0.25f * m * Wr[d];
  }
#pragma unroll
  for (int off = 16; off > 0; off >>= 1) p += __shfl_down(p, off);
  if (lane == 0) atomicAdd(out, p + br[0]);
}

extern "C" void kernel_launch(void* const* d_in, const int* in_sizes, int n_in,
                              void* d_out, int out_size, void* d_ws, size_t ws_size,
                              hipStream_t stream) {
  const float* feats = (const float*)d_in[0];
  const int*   src   = (const int*)d_in[1];
  const int*   dst   = (const int*)d_in[2];
  const float* W1    = (const float*)d_in[3];
  const float* al1   = (const float*)d_in[4];
  const float* ar1   = (const float*)d_in[5];
  const float* b1    = (const float*)d_in[6];
  const float* W2    = (const float*)d_in[7];
  const float* al2   = (const float*)d_in[8];
  const float* ar2   = (const float*)d_in[9];
  const float* b2    = (const float*)d_in[10];
  const float* Wr    = (const float*)d_in[11];
  const float* br    = (const float*)d_in[12];

  // workspace layout (all 16B aligned): ~89 MB
  float* Z   = (float*)d_ws;                         // [N, CH] activations (z)
  float* ACC = Z   + (size_t)N_NODES * CH;           // [N, CH] aggregation / h1
  float* EA  = ACC + (size_t)N_NODES * CH;           // [E, H]  edge scores/weights
  float* EL  = EA  + (size_t)N_EDGES * HEADS;        // [N, H]
  float* ER  = EL  + (size_t)N_NODES * HEADS;        // [N, H]
  float* EM  = ER  + (size_t)N_NODES * HEADS;        // [N, H]  segment max
  float* DEN = EM  + (size_t)N_NODES * HEADS;        // [N, H]  softmax denom
  _Float16* WT = (_Float16*)(DEN + (size_t)N_NODES * HEADS); // [CH, K] f16 W^T

  const int T  = 256;                    // 8 waves per block
  const int nh = N_NODES * HEADS;
  const int eh = N_EDGES * HEADS;
  const int gemm_waves    = (N_NODES / 16) * (CH / 64);
  const int gemm_blocks   = (gemm_waves + 7) / 8;
  const int score_blocks  = (nh + 7) / 8;            // one wave per (node,head)
  const int scatter_blocks = (N_EDGES + 7) / 8;      // one wave per edge
  const int readout_blocks = (N_NODES + 7) / 8;      // one wave per node

  // ================= layer 1 =================
  wt_conv_kernel<<<(F_IN * CH + T - 1) / T, T, 0, stream>>>(W1, WT, F_IN, CH);
  gemm_wmma_kernel<<<gemm_blocks, T, 0, stream>>>(feats, WT, Z, N_NODES, CH, F_IN);
  attn_score_kernel<<<score_blocks, T, 0, stream>>>(Z, al1, ar1, EL, ER);
  fill_kernel<<<(nh + T - 1) / T, T, 0, stream>>>(EM, -INFINITY, nh);
  hipMemsetAsync(DEN, 0, (size_t)nh * sizeof(float), stream);
  hipMemsetAsync(ACC, 0, (size_t)N_NODES * CH * sizeof(float), stream);
  edge_max_kernel<<<(eh + T - 1) / T, T, 0, stream>>>(src, dst, EL, ER, EA, EM);
  edge_exp_kernel<<<(eh + T - 1) / T, T, 0, stream>>>(dst, EA, EM, DEN);
  edge_scatter_kernel<<<scatter_blocks, T, 0, stream>>>(src, dst, EA, DEN, Z, ACC);
  bias_elu_kernel<<<(N_NODES * CH + T - 1) / T, T, 0, stream>>>(ACC, b1); // ACC := h1

  // ================= layer 2 =================
  wt_conv_kernel<<<(CH * CH + T - 1) / T, T, 0, stream>>>(W2, WT, CH, CH);
  gemm_wmma_kernel<<<gemm_blocks, T, 0, stream>>>(ACC, WT, Z, N_NODES, CH, CH);
  attn_score_kernel<<<score_blocks, T, 0, stream>>>(Z, al2, ar2, EL, ER);
  fill_kernel<<<(nh + T - 1) / T, T, 0, stream>>>(EM, -INFINITY, nh);
  hipMemsetAsync(DEN, 0, (size_t)nh * sizeof(float), stream);
  hipMemsetAsync(ACC, 0, (size_t)N_NODES * CH * sizeof(float), stream);
  edge_max_kernel<<<(eh + T - 1) / T, T, 0, stream>>>(src, dst, EL, ER, EA, EM);
  edge_exp_kernel<<<(eh + T - 1) / T, T, 0, stream>>>(dst, EA, EM, DEN);
  edge_scatter_kernel<<<scatter_blocks, T, 0, stream>>>(src, dst, EA, DEN, Z, ACC);

  // ================= readout =================
  hipMemsetAsync(d_out, 0, sizeof(float), stream);
  readout_kernel<<<readout_blocks, T, 0, stream>>>(ACC, b2, Wr, br, (float*)d_out);
}
